// Qwen3MoeDecoderLayer_39256001086126
// MI455X (gfx1250) — matrix operand order repaired and checked by reading, not compiled, not run on hardware
//
#include <hip/hip_runtime.h>
#include <hip/hip_bf16.h>

// ---------------- model constants ----------------
#define T_TOK 1024
#define H     2048
#define NH    16
#define NKV   4
#define D     128
#define E     32
#define TOPK  8
#define F     768
#define QKVW  ((NH + 2 * NKV) * D)   // 3072
#define EPSF  1e-6f
#define THETA 10000.0f

// ---------------- WMMA helpers (gfx1250, wave32) ----------------
typedef __attribute__((ext_vector_type(16))) __bf16 v16bf;
typedef __attribute__((ext_vector_type(8)))  float  v8f;
typedef __attribute__((ext_vector_type(4)))  __bf16 v4bf;
typedef __attribute__((ext_vector_type(2)))  __bf16 v2bf;

__device__ inline v8f zero8() { v8f z = {}; return z; }

__device__ inline v8f wmma_bf16(v16bf a, v16bf b, v8f c) {
  return __builtin_amdgcn_wmma_f32_16x16x32_bf16(false, a, false, b, (short)0, c,
                                                 false, false);
}

// A-tile k layout: identity, so each lane's 16 elements are contiguous
// (lanes 0-15 hold K=0..15; lanes 16-31 hold
// K=16..31), matching the B-tile layout.
__device__ inline int aswz(int k) {
  return k;
}
// B-tile swizzle: subtile = n>>4 (512 elems each); within a subtile lane
// (khalf*16 + n&15) owns 16 contiguous elements indexed by k&15.
__device__ inline int bswz(int k, int n) {
  return ((n >> 4) << 9) + (((((k >> 4) << 4) | (n & 15))) << 4) + (k & 15);
}

// A fragment from swizzled A tile (row-major, ld=32): one contiguous v16bf.
__device__ inline v16bf frag_a(const __bf16* As, int row0) {
  const int lane = threadIdx.x & 31;
  return *(const v16bf*)(As + (row0 + (lane & 15)) * 32 + ((lane >> 4) << 4));
}
// B fragment from swizzled B tile: lane l reads its own 32B chunk.
__device__ inline v16bf frag_b(const __bf16* Bs, int sub) {
  const int lane = threadIdx.x & 31;
  return ((const v16bf*)(Bs + (sub << 9)))[lane];
}

// ---------------- RMSNorm ----------------
__global__ __launch_bounds__(256) void k_rmsnorm(const float* __restrict__ x,
                                                 const float* __restrict__ w,
                                                 float* __restrict__ out) {
  const int t = blockIdx.x;
  const float* xr = x + (size_t)t * H;
  __shared__ float red[8];
  float ss = 0.f;
  for (int i = threadIdx.x; i < H; i += 256) { float v = xr[i]; ss += v * v; }
  for (int off = 16; off > 0; off >>= 1) ss += __shfl_xor(ss, off);
  if ((threadIdx.x & 31) == 0) red[threadIdx.x >> 5] = ss;
  __syncthreads();
  float tot = 0.f;
#pragma unroll
  for (int i = 0; i < 8; ++i) tot += red[i];
  const float rstd = rsqrtf(tot / (float)H + EPSF);
  float* orow = out + (size_t)t * H;
  for (int i = threadIdx.x; i < H; i += 256) orow[i] = xr[i] * rstd * w[i];
}

// ---------------- staging helpers (fp32 global -> bf16 LDS) ----------------
#define BK  32
#define GBM 128
#define GBN 128
#define MBM 64
#define MBN 64

// A tile: ROWS x 32, swizzled; float4 global loads, ds_store_b64 stores.
template <int ROWS, bool BOUND>
__device__ inline void stage_a(__bf16* As, const float* A, int m0, int k0,
                               int M, int K) {
  for (int g = threadIdx.x; g < (ROWS * BK) / 4; g += 256) {
    const int r = g >> 3, c4 = (g & 7) << 2;
    const int gr = m0 + r;
    float x0 = 0.f, x1 = 0.f, x2 = 0.f, x3 = 0.f;
    if (!BOUND || gr < M) {
      const float4 v = *(const float4*)(A + (size_t)gr * K + k0 + c4);
      x0 = v.x; x1 = v.y; x2 = v.z; x3 = v.w;
    }
    v4bf o;
    o[0] = (__bf16)x0; o[1] = (__bf16)x1; o[2] = (__bf16)x2; o[3] = (__bf16)x3;
    *(v4bf*)(As + r * 32 + aswz(c4)) = o;
  }
}

// A tile with row gather (MoE): token list indirection.
template <int ROWS>
__device__ inline void stage_a_gather(__bf16* As, const float* A, int m0, int k0,
                                      int cnt, int K, const int* rows) {
  for (int g = threadIdx.x; g < (ROWS * BK) / 4; g += 256) {
    const int r = g >> 3, c4 = (g & 7) << 2;
    const int slot = m0 + r;
    float x0 = 0.f, x1 = 0.f, x2 = 0.f, x3 = 0.f;
    if (slot < cnt) {
      const float4 v = *(const float4*)(A + (size_t)rows[slot] * K + k0 + c4);
      x0 = v.x; x1 = v.y; x2 = v.z; x3 = v.w;
    }
    v4bf o;
    o[0] = (__bf16)x0; o[1] = (__bf16)x1; o[2] = (__bf16)x2; o[3] = (__bf16)x3;
    *(v4bf*)(As + r * 32 + aswz(c4)) = o;
  }
}

// B tile 32xBNT swizzled: 2 float4 row loads, 4 packed-pair ds_store_b32.
template <int BNT>
__device__ inline void stage_b(__bf16* Bs, const float* B, int n0, int k0, int N) {
  constexpr int NG = BNT / 4;
  for (int g = threadIdx.x; g < (BK * BNT) / 8; g += 256) {
    const int n = (g % NG) << 2;
    const int k = (g / NG) << 1;
    const float* p = B + (size_t)(k0 + k) * N + n0 + n;
    const float4 r0 = *(const float4*)p;
    const float4 r1 = *(const float4*)(p + N);
    const float* a0 = (const float*)&r0;
    const float* a1 = (const float*)&r1;
#pragma unroll
    for (int j = 0; j < 4; ++j) {
      v2bf t; t[0] = (__bf16)a0[j]; t[1] = (__bf16)a1[j];
      *(v2bf*)(Bs + bswz(k, n + j)) = t;
    }
  }
}

// ---------------- generic GEMM: C[M,N] = A[M,K] @ B[K,N] (+resid) ----------------
// 128x128 block, 8 waves as 4x2, each wave 32x64 (2x4 WMMA tiles).
__global__ __launch_bounds__(256) void k_gemm(const float* __restrict__ A,
                                              const float* __restrict__ B,
                                              float* __restrict__ C,
                                              const float* __restrict__ resid,
                                              int M, int N, int K) {
  __shared__ __bf16 As[2][GBM * BK];
  __shared__ __bf16 Bs[2][BK * GBN];
  const int m0 = blockIdx.y * GBM;
  const int n0 = blockIdx.x * GBN;
  const int wave = threadIdx.x >> 5;
  const int wm = wave & 3;   // 4 wave rows x 32
  const int wn = wave >> 2;  // 2 wave cols x 64
  v8f acc[2][4];
#pragma unroll
  for (int i = 0; i < 2; ++i)
#pragma unroll
    for (int j = 0; j < 4; ++j) acc[i][j] = zero8();

  stage_a<GBM, false>(As[0], A, m0, 0, M, K);
  stage_b<GBN>(Bs[0], B, n0, 0, N);
  int buf = 0;
  for (int k0 = 0; k0 < K; k0 += BK) {
    __syncthreads();
    if (k0 + BK < K) {
      stage_a<GBM, false>(As[buf ^ 1], A, m0, k0 + BK, M, K);
      stage_b<GBN>(Bs[buf ^ 1], B, n0, k0 + BK, N);
    }
    const v16bf a0 = frag_a(As[buf], wm * 32);
    const v16bf a1 = frag_a(As[buf], wm * 32 + 16);
#pragma unroll
    for (int sn = 0; sn < 4; ++sn) {
      const v16bf b = frag_b(Bs[buf], wn * 4 + sn);
      acc[0][sn] = wmma_bf16(a0, b, acc[0][sn]);
      acc[1][sn] = wmma_bf16(a1, b, acc[1][sn]);
    }
    buf ^= 1;
  }
  const int lane = threadIdx.x & 31, nl = lane & 15, mb = (lane >> 4) << 3;
  if (resid) {
#pragma unroll
    for (int sm = 0; sm < 2; ++sm)
#pragma unroll
      for (int sn = 0; sn < 4; ++sn) {
        const int cg = n0 + wn * 64 + sn * 16 + nl;
#pragma unroll
        for (int j = 0; j < 8; ++j) {
          const int rg = m0 + wm * 32 + sm * 16 + mb + j;
          const size_t o = (size_t)rg * N + cg;
          C[o] = resid[o] + acc[sm][sn][j];
        }
      }
  } else {
#pragma unroll
    for (int sm = 0; sm < 2; ++sm)
#pragma unroll
      for (int sn = 0; sn < 4; ++sn) {
        const int cg = n0 + wn * 64 + sn * 16 + nl;
#pragma unroll
        for (int j = 0; j < 8; ++j) {
          const int rg = m0 + wm * 32 + sm * 16 + mb + j;
          C[(size_t)rg * N + cg] = acc[sm][sn][j];
        }
      }
  }
}

// ---------------- fused q/k RMSNorm + RoPE (in place on qkv) ----------------
__global__ __launch_bounds__(256) void k_qknorm_rope(float* __restrict__ qkv,
                                                     const int* __restrict__ positions,
                                                     const float* __restrict__ qw,
                                                     const float* __restrict__ kw) {
  const int t = blockIdx.x;
  const int wave = threadIdx.x >> 5;
  const int lane = threadIdx.x & 31;
  const float pos = (float)positions[t];
  for (int s = wave; s < NH + NKV; s += 8) {
    const float* w = (s < NH) ? qw : kw;
    float* x = qkv + (size_t)t * QKVW + s * D;
    float v0 = x[lane], v1 = x[lane + 32], v2 = x[lane + 64], v3 = x[lane + 96];
    float ss = v0 * v0 + v1 * v1 + v2 * v2 + v3 * v3;
    for (int off = 16; off > 0; off >>= 1) ss += __shfl_xor(ss, off);
    const float rstd = rsqrtf(ss / (float)D + EPSF);
    v0 *= rstd * w[lane];      v1 *= rstd * w[lane + 32];
    v2 *= rstd * w[lane + 64]; v3 *= rstd * w[lane + 96];
    const float f0 = powf(THETA, -((float)lane) / 64.f);
    const float f1 = powf(THETA, -((float)(lane + 32)) / 64.f);
    const float c0 = cosf(pos * f0), s0 = sinf(pos * f0);
    const float c1 = cosf(pos * f1), s1 = sinf(pos * f1);
    x[lane]      = v0 * c0 - v2 * s0;
    x[lane + 64] = v2 * c0 + v0 * s0;
    x[lane + 32] = v1 * c1 - v3 * s1;
    x[lane + 96] = v3 * c1 + v1 * s1;
  }
}

// ---------------- flash attention (causal, GQA 4:1) ----------------
__global__ __launch_bounds__(256) void k_attn(const float* __restrict__ qkv,
                                              float* __restrict__ out) {
  const int h = blockIdx.y;
  const int kvh = h >> 2;
  const int wave = threadIdx.x >> 5;
  const int lane = threadIdx.x & 31;
  const int qrow0 = blockIdx.x * 128 + wave * 16;
  __shared__ __bf16 Ks[32 * 128];      // row-major [s][d]
  __shared__ __bf16 Vs[32 * 128];      // B-swizzled by (k=s, n=d)
  __shared__ __bf16 Ps[8][16 * 32];    // per-wave, A-swizzled
  const float scale = 0.08838834764831845f;  // D^-0.5
  const int m = lane & 15;
  const int kb8  = (lane >> 4) << 3;
  const int kb16 = (lane >> 4) << 4;
  const int nl = m, mb = kb8;

  // preload scaled Q fragments (4 k-steps of 32 over D=128), float4 loads
  v16bf qf[4];
  {
    const float* qb = qkv + (size_t)(qrow0 + m) * QKVW + h * D;
#pragma unroll
    for (int kk = 0; kk < 4; ++kk) {
      const float4 q0 = *(const float4*)(qb + kk * 32 + kb16);
      const float4 q1 = *(const float4*)(qb + kk * 32 + kb16 + 4);
      const float4 q2 = *(const float4*)(qb + kk * 32 + kb16 + 8);
      const float4 q3 = *(const float4*)(qb + kk * 32 + kb16 + 12);
      const float* f0 = (const float*)&q0;
      const float* f1 = (const float*)&q1;
      const float* f2 = (const float*)&q2;
      const float* f3 = (const float*)&q3;
      v16bf a;
#pragma unroll
      for (int i = 0; i < 4; ++i) {
        a[i]      = (__bf16)(f0[i] * scale);
        a[i + 4]  = (__bf16)(f1[i] * scale);
        a[i + 8]  = (__bf16)(f2[i] * scale);
        a[i + 12] = (__bf16)(f3[i] * scale);
      }
      qf[kk] = a;
    }
  }
  float mrow[8], lrow[8];
  v8f o[8];
#pragma unroll
  for (int j = 0; j < 8; ++j) { mrow[j] = -1e30f; lrow[j] = 0.f; }
#pragma unroll
  for (int n = 0; n < 8; ++n) o[n] = zero8();

  const int ntile = (blockIdx.x * 128 + 127) / 32 + 1;
  for (int ti = 0; ti < ntile; ++ti) {
    const int s0 = ti * 32;
    // K: row-major, vectorized
    for (int g = threadIdx.x; g < (32 * 128) / 4; g += 256) {
      const int r = g >> 5, c4 = (g & 31) << 2;
      const float4 v = *(const float4*)(qkv + (size_t)(s0 + r) * QKVW +
                                        (NH + kvh) * D + c4);
      v4bf t;
      t[0] = (__bf16)v.x; t[1] = (__bf16)v.y; t[2] = (__bf16)v.z; t[3] = (__bf16)v.w;
      *(v4bf*)(Ks + r * 128 + c4) = t;
    }
    // V: B-swizzled, packed s-pairs
    for (int g = threadIdx.x; g < (32 * 128) / 8; g += 256) {
      const int d = (g & 31) << 2;
      const int s = (g >> 5) << 1;
      const float* p = qkv + (size_t)(s0 + s) * QKVW + (NH + NKV + kvh) * D + d;
      const float4 r0 = *(const float4*)p;
      const float4 r1 = *(const float4*)(p + QKVW);
      const float* a0 = (const float*)&r0;
      const float* a1 = (const float*)&r1;
#pragma unroll
      for (int j = 0; j < 4; ++j) {
        v2bf t; t[0] = (__bf16)a0[j]; t[1] = (__bf16)a1[j];
        *(v2bf*)(Vs + bswz(s, d + j)) = t;
      }
    }
    __syncthreads();
    const bool active = (s0 <= qrow0 + 15);
    if (active) {
      v8f sc[2]; sc[0] = zero8(); sc[1] = zero8();
#pragma unroll
      for (int kk = 0; kk < 4; ++kk)
#pragma unroll
        for (int f = 0; f < 2; ++f) {
          // B = K^T: lane reads 16 contiguous d-values of one key row
          const v16bf b = *(const v16bf*)(Ks + (f * 16 + nl) * 128 + kk * 32 + kb16);
          sc[f] = wmma_bf16(qf[kk], b, sc[f]);
        }
      // causal mask + online softmax (rowwise across 16-lane halves)
#pragma unroll
      for (int j = 0; j < 8; ++j) {
        const int row = qrow0 + mb + j;
        float x0 = sc[0][j], x1 = sc[1][j];
        if (s0 + nl > row)      x0 = -1e30f;
        if (s0 + 16 + nl > row) x1 = -1e30f;
        float rmax = fmaxf(x0, x1);
        for (int off = 1; off < 16; off <<= 1) rmax = fmaxf(rmax, __shfl_xor(rmax, off));
        const float mnew = fmaxf(mrow[j], rmax);
        const float p0 = expf(x0 - mnew), p1 = expf(x1 - mnew);
        float rsum = p0 + p1;
        for (int off = 1; off < 16; off <<= 1) rsum += __shfl_xor(rsum, off);
        const float lsc = expf(mrow[j] - mnew);
        lrow[j] = lrow[j] * lsc + rsum;
        mrow[j] = mnew;
#pragma unroll
        for (int n = 0; n < 8; ++n) o[n][j] *= lsc;
        sc[0][j] = p0; sc[1][j] = p1;
      }
      // stage P into A-swizzled layout for the PV matmul
      __bf16* pw = &Ps[wave][0];
#pragma unroll
      for (int f = 0; f < 2; ++f)
#pragma unroll
        for (int j = 0; j < 8; ++j)
          pw[(mb + j) * 32 + aswz(f * 16 + nl)] = (__bf16)sc[f][j];
    }
    __syncthreads();
    if (active) {
      const v16bf a = frag_a(&Ps[wave][0], 0);
#pragma unroll
      for (int n = 0; n < 8; ++n) {
        const v16bf b = frag_b(Vs, n);
        o[n] = wmma_bf16(a, b, o[n]);
      }
    }
    __syncthreads();
  }
#pragma unroll
  for (int n = 0; n < 8; ++n)
#pragma unroll
    for (int j = 0; j < 8; ++j) {
      const int row = qrow0 + mb + j;
      out[(size_t)row * (NH * D) + h * D + n * 16 + nl] = o[n][j] / lrow[j];
    }
}

// ---------------- router: softmax + top-8 per token (one wave / token) ----------------
__global__ __launch_bounds__(32) void k_router(const float* __restrict__ h2,
                                               const float* __restrict__ gate_w,
                                               float* __restrict__ comb) {
  const int t = blockIdx.x;
  const int e = threadIdx.x;
  const float* hr = h2 + (size_t)t * H;
  float acc = 0.f;
  for (int hh = 0; hh < H; ++hh) acc += hr[hh] * gate_w[hh * E + e];
  float mx = acc;
  for (int off = 16; off > 0; off >>= 1) mx = fmaxf(mx, __shfl_xor(mx, off));
  float p = expf(acc - mx);
  float sm = p;
  for (int off = 16; off > 0; off >>= 1) sm += __shfl_xor(sm, off);
  p /= sm;
  float cur = p, keep = 0.f, sumtop = 0.f;
  for (int it = 0; it < TOPK; ++it) {
    float m = cur;
    for (int off = 16; off > 0; off >>= 1) m = fmaxf(m, __shfl_xor(m, off));
    const unsigned long long bal = __ballot(cur == m);
    const int sel = __ffsll((long long)bal) - 1;   // lowest lane wins ties
    sumtop += m;
    if (e == sel) { keep = p; cur = -1.f; }
  }
  comb[t * E + e] = keep / sumtop;
}

// ---------------- deterministic per-expert token gather ----------------
__global__ void k_lists(const float* __restrict__ comb, int* __restrict__ counts,
                        int* __restrict__ tlist, int* __restrict__ slot_idx) {
  const int e = blockIdx.x;
  if (threadIdx.x != 0) return;
  int cnt = 0;
  for (int t = 0; t < T_TOK; ++t) {
    if (comb[t * E + e] > 0.f) {
      tlist[e * T_TOK + cnt] = t;
      slot_idx[e * T_TOK + t] = cnt;
      ++cnt;
    } else {
      slot_idx[e * T_TOK + t] = -1;
    }
  }
  counts[e] = cnt;
}

__global__ void k_offsets(const int* __restrict__ counts, int* __restrict__ offsets) {
  if (threadIdx.x != 0 || blockIdx.x != 0) return;
  int run = 0;
  for (int e = 0; e < E; ++e) { offsets[e] = run; run += counts[e]; }
}

// ---------------- MoE gate_up GEMM + SiLU*up (gathered rows) ----------------
__global__ __launch_bounds__(256) void k_moe_gate_up(const float* __restrict__ h2,
                                                     const float* __restrict__ wgu,
                                                     const int* __restrict__ counts,
                                                     const int* __restrict__ offsets,
                                                     const int* __restrict__ tlist,
                                                     float* __restrict__ act) {
  const int e = blockIdx.z;
  const int cnt = counts[e];
  const int m0 = blockIdx.y * MBM;
  if (m0 >= cnt) return;
  const int n0 = blockIdx.x * MBN;  // within F
  __shared__ __bf16 As[2][MBM * BK];
  __shared__ __bf16 Bg[2][BK * MBN];
  __shared__ __bf16 Bu[2][BK * MBN];
  const int wave = threadIdx.x >> 5;
  const int wm = wave >> 1;  // 4 wave rows x 16
  const int wn = wave & 1;   // 2 wave cols x 32
  const float* wB = wgu + (size_t)e * H * (2 * F);
  const int* rows = tlist + e * T_TOK;
  v8f ag[2], au[2];
  ag[0] = zero8(); ag[1] = zero8(); au[0] = zero8(); au[1] = zero8();

  stage_a_gather<MBM>(As[0], h2, m0, 0, cnt, H, rows);
  stage_b<MBN>(Bg[0], wB, n0, 0, 2 * F);
  stage_b<MBN>(Bu[0], wB, n0 + F, 0, 2 * F);
  int buf = 0;
  for (int k0 = 0; k0 < H; k0 += BK) {
    __syncthreads();
    if (k0 + BK < H) {
      stage_a_gather<MBM>(As[buf ^ 1], h2, m0, k0 + BK, cnt, H, rows);
      stage_b<MBN>(Bg[buf ^ 1], wB, n0, k0 + BK, 2 * F);
      stage_b<MBN>(Bu[buf ^ 1], wB, n0 + F, k0 + BK, 2 * F);
    }
    const v16bf a = frag_a(As[buf], wm * 16);
#pragma unroll
    for (int sn = 0; sn < 2; ++sn) {
      ag[sn] = wmma_bf16(a, frag_b(Bg[buf], wn * 2 + sn), ag[sn]);
      au[sn] = wmma_bf16(a, frag_b(Bu[buf], wn * 2 + sn), au[sn]);
    }
    buf ^= 1;
  }
  const int lane = threadIdx.x & 31, nl = lane & 15, mb = (lane >> 4) << 3;
  const int ob = offsets[e];
#pragma unroll
  for (int sn = 0; sn < 2; ++sn) {
    const int f = n0 + wn * 32 + sn * 16 + nl;
#pragma unroll
    for (int j = 0; j < 8; ++j) {
      const int slot = m0 + wm * 16 + mb + j;
      if (slot < cnt) {
        const float g = ag[sn][j], u = au[sn][j];
        act[(size_t)(ob + slot) * F + f] = (g / (1.f + expf(-g))) * u;
      }
    }
  }
}

// ---------------- MoE down GEMM (packed rows, unscaled) ----------------
__global__ __launch_bounds__(256) void k_moe_down(const float* __restrict__ act,
                                                  const float* __restrict__ wdn,
                                                  const int* __restrict__ counts,
                                                  const int* __restrict__ offsets,
                                                  float* __restrict__ eo) {
  const int e = blockIdx.z;
  const int cnt = counts[e];
  const int m0 = blockIdx.y * MBM;
  if (m0 >= cnt) return;
  const int n0 = blockIdx.x * MBN;  // within H
  __shared__ __bf16 As[2][MBM * BK];
  __shared__ __bf16 Bs[2][BK * MBN];
  const int wave = threadIdx.x >> 5;
  const int wm = wave >> 1;
  const int wn = wave & 1;
  const int ob = offsets[e];
  const float* Ae = act + (size_t)ob * F;
  const float* Be = wdn + (size_t)e * F * H;
  v8f acc[2]; acc[0] = zero8(); acc[1] = zero8();

  stage_a<MBM, true>(As[0], Ae, m0, 0, cnt, F);
  stage_b<MBN>(Bs[0], Be, n0, 0, H);
  int buf = 0;
  for (int k0 = 0; k0 < F; k0 += BK) {
    __syncthreads();
    if (k0 + BK < F) {
      stage_a<MBM, true>(As[buf ^ 1], Ae, m0, k0 + BK, cnt, F);
      stage_b<MBN>(Bs[buf ^ 1], Be, n0, k0 + BK, H);
    }
    const v16bf a = frag_a(As[buf], wm * 16);
#pragma unroll
    for (int sn = 0; sn < 2; ++sn)
      acc[sn] = wmma_bf16(a, frag_b(Bs[buf], wn * 2 + sn), acc[sn]);
    buf ^= 1;
  }
  const int lane = threadIdx.x & 31, nl = lane & 15, mb = (lane >> 4) << 3;
#pragma unroll
  for (int sn = 0; sn < 2; ++sn) {
    const int hcol = n0 + wn * 32 + sn * 16 + nl;
#pragma unroll
    for (int j = 0; j < 8; ++j) {
      const int slot = m0 + wm * 16 + mb + j;
      if (slot < cnt) eo[(size_t)(ob + slot) * H + hcol] = acc[sn][j];
    }
  }
}

// ---------------- deterministic combine: ascending expert order ----------------
__global__ __launch_bounds__(256) void k_moe_combine(const float* __restrict__ eo,
                                                     const float* __restrict__ comb,
                                                     const int* __restrict__ offsets,
                                                     const int* __restrict__ slot_idx,
                                                     float* __restrict__ moe_out) {
  const int t = blockIdx.x;
  const int c0 = threadIdx.x * 8;
  float a[8];
#pragma unroll
  for (int i = 0; i < 8; ++i) a[i] = 0.f;
  for (int e = 0; e < E; ++e) {
    const float cw = comb[t * E + e];
    if (cw > 0.f) {
      const int s = slot_idx[e * T_TOK + t];
      const float* row = eo + (size_t)(offsets[e] + s) * H;
#pragma unroll
      for (int i = 0; i < 8; ++i) a[i] += cw * row[c0 + i];
    }
  }
#pragma unroll
  for (int i = 0; i < 8; ++i) moe_out[(size_t)t * H + c0 + i] = a[i];
}

// ---------------- host launcher ----------------
extern "C" void kernel_launch(void* const* d_in, const int* in_sizes, int n_in,
                              void* d_out, int out_size, void* d_ws, size_t ws_size,
                              hipStream_t stream) {
  const float* hidden    = (const float*)d_in[0];
  const int*   positions = (const int*)  d_in[1];
  const float* qkv_w     = (const float*)d_in[2];
  const float* o_w       = (const float*)d_in[3];
  const float* q_norm_w  = (const float*)d_in[4];
  const float* k_norm_w  = (const float*)d_in[5];
  const float* ln1_w     = (const float*)d_in[6];
  const float* ln2_w     = (const float*)d_in[7];
  const float* gate_w    = (const float*)d_in[8];
  const float* w_gate_up = (const float*)d_in[9];
  const float* w_down    = (const float*)d_in[10];
  (void)in_sizes; (void)n_in; (void)out_size; (void)ws_size;

  float* moe_out  = (float*)d_out;
  float* residual = moe_out + (size_t)T_TOK * H;  // second tuple output

  float* ws = (float*)d_ws;
  float* h1   = ws;  ws += (size_t)T_TOK * H;
  float* qkv  = ws;  ws += (size_t)T_TOK * QKVW;
  float* attn = ws;  ws += (size_t)T_TOK * NH * D;
  float* h2   = ws;  ws += (size_t)T_TOK * H;
  float* comb = ws;  ws += (size_t)T_TOK * E;
  float* act  = ws;  ws += (size_t)T_TOK * TOPK * F;   // 8192 x 768
  float* eo   = ws;  ws += (size_t)T_TOK * TOPK * H;   // 8192 x 2048
  int* counts  = (int*)ws;
  int* offsets = counts + E;
  int* tlist   = offsets + E;        // E x T
  int* slotidx = tlist + E * T_TOK;  // E x T

  k_rmsnorm<<<T_TOK, 256, 0, stream>>>(hidden, ln1_w, h1);
  k_gemm<<<dim3(QKVW / GBN, T_TOK / GBM), 256, 0, stream>>>(h1, qkv_w, qkv, nullptr,
                                                            T_TOK, QKVW, H);
  k_qknorm_rope<<<T_TOK, 256, 0, stream>>>(qkv, positions, q_norm_w, k_norm_w);
  k_attn<<<dim3(T_TOK / 128, NH), 256, 0, stream>>>(qkv, attn);
  k_gemm<<<dim3(H / GBN, T_TOK / GBM), 256, 0, stream>>>(attn, o_w, residual, hidden,
                                                         T_TOK, H, NH * D);
  k_rmsnorm<<<T_TOK, 256, 0, stream>>>(residual, ln2_w, h2);
  k_router<<<T_TOK, 32, 0, stream>>>(h2, gate_w, comb);
  k_lists<<<E, 32, 0, stream>>>(comb, counts, tlist, slotidx);
  k_offsets<<<1, 32, 0, stream>>>(counts, offsets);
  k_moe_gate_up<<<dim3(F / MBN, T_TOK / MBM, E), 256, 0, stream>>>(h2, w_gate_up, counts,
                                                                   offsets, tlist, act);
  k_moe_down<<<dim3(H / MBN, T_TOK / MBM, E), 256, 0, stream>>>(act, w_down, counts,
                                                                offsets, eo);
  k_moe_combine<<<T_TOK, 256, 0, stream>>>(eo, comb, offsets, slotidx, moe_out);
}
